// CrossAttention_67568425501540
// MI455X (gfx1250) — matrix operand order repaired and checked by reading, not compiled, not run on hardware
//
#include <hip/hip_runtime.h>

// ---------------------------------------------------------------------------
// CDNA5 (gfx1250) cross-attention block, f16 WMMA everywhere, async-to-LDS
// double-buffered staging (GLOBAL_LOAD_ASYNC_TO_LDS_B128 + S_WAIT_ASYNCCNT).
//   DIM=1024, HEADS=16, DIM_HEAD=64, B=4, N=2048, SCALE=1/8
// Pipeline:
//   1. layernorm + cast x -> f16          (2 launches)
//   2. cast weights -> f16                (4 launches)
//   3. QKV GEMM  [8192x1024]x[1024x3072]  (2 launches, WMMA, f16 out)
//   4. flash attention, online softmax    (2 launches, WMMA QK^T and PV)
//   5. out-proj GEMM + bias, f32 out      (2 launches, WMMA)
// ---------------------------------------------------------------------------

typedef __attribute__((ext_vector_type(16))) _Float16 v16h;
typedef __attribute__((ext_vector_type(8)))  float    v8f;

#define DIM       1024
#define HEADS     16
#define DIM_HEAD  64
#define NSEQ      2048
#define BATCH     4
#define MROWS     (BATCH * NSEQ)        // 8192
#define NQKV      (3 * DIM)             // 3072
#define ATT_SCALE 0.125f

#define LDSP 80   // LDS tile row stride in halves (160B: 16B aligned, avoids 64-bank conflicts)

__device__ __forceinline__ v8f wmma_f16(v16h a, v16h b, v8f c) {
  // D = A(16x32 f16) * B(32x16 f16) + C(16x16 f32)
  return __builtin_amdgcn_wmma_f32_16x16x32_f16(false, a, false, b,
                                                (short)0, c, false, false);
}

// Async 16B global -> LDS copy (ASYNCcnt-tracked, no VGPR round trip).
// Generic LDS address carries the LDS byte offset in its low 32 bits.
__device__ __forceinline__ void async_copy16(void* lds_dst, const void* gsrc) {
  asm volatile("global_load_async_to_lds_b128 %0, %1, off"
               :: "v"((unsigned)(uintptr_t)lds_dst),
                  "v"((unsigned long long)(uintptr_t)gsrc)
               : "memory");
}
__device__ __forceinline__ void async_wait0() {
  asm volatile("s_wait_asynccnt 0x0" ::: "memory");
}

// A fragment (16x32, MxK): lane 0-15 row M=lane, K=0..15;
// lanes 16-31 same rows, K=16..31 (matches B fragment K order)
__device__ __forceinline__ v16h load_a_frag(const _Float16* p, int ld, int lane) {
  const int m   = lane & 15;
  const int khi = (lane >> 4) << 4;
  v16h a;
#pragma unroll
  for (int i = 0; i < 16; ++i) {
    const int k = khi + i;
    a[i] = p[m * ld + k];
  }
  return a;
}

// B fragment (32x16, KxN), source stored row-major [k][n]:
// lane 0-15 col N=lane K=0..15; lanes 16-31 col N=lane-16 K=16..31
__device__ __forceinline__ v16h load_b_frag(const _Float16* p, int ld, int lane) {
  const int n  = lane & 15;
  const int kb = (lane >> 4) << 4;
  v16h b;
#pragma unroll
  for (int i = 0; i < 16; ++i) b[i] = p[(kb + i) * ld + n];
  return b;
}

// B fragment gathered from a transposed source ([n][k] in LDS) -> free K^T
__device__ __forceinline__ v16h load_bT_frag(const _Float16* p, int ld, int lane) {
  const int n  = lane & 15;
  const int kb = (lane >> 4) << 4;
  v16h b;
#pragma unroll
  for (int i = 0; i < 16; ++i) b[i] = p[n * ld + kb + i];
  return b;
}

// ---------------------------------------------------------------------------
// 1. LayerNorm + cast to f16. One row (1024 f32) per 256-thread block.
// ---------------------------------------------------------------------------
__global__ __launch_bounds__(256) void ln_cast_kernel(
    const float* __restrict__ x, const float* __restrict__ gamma,
    const float* __restrict__ beta, _Float16* __restrict__ out) {
  const int row = blockIdx.x;
  const int t   = threadIdx.x;
  const float* xr = x + (size_t)row * DIM;

  const float4 f4 = *(const float4*)(xr + t * 4);
  float vals[4] = {f4.x, f4.y, f4.z, f4.w};
  float s = 0.f, ss = 0.f;
#pragma unroll
  for (int i = 0; i < 4; ++i) { s += vals[i]; ss += vals[i] * vals[i]; }
#pragma unroll
  for (int m = 16; m >= 1; m >>= 1) {       // wave32 reduction
    s  += __shfl_xor(s,  m, 32);
    ss += __shfl_xor(ss, m, 32);
  }
  __shared__ float sums[8], sqs[8];
  if ((t & 31) == 0) { sums[t >> 5] = s; sqs[t >> 5] = ss; }
  __syncthreads();
  float S = 0.f, SS = 0.f;
#pragma unroll
  for (int w = 0; w < 8; ++w) { S += sums[w]; SS += sqs[w]; }
  const float mu   = S * (1.0f / DIM);
  const float rstd = rsqrtf(SS * (1.0f / DIM) - mu * mu + 1e-5f);

  _Float16* orow = out + (size_t)row * DIM;
#pragma unroll
  for (int i = 0; i < 4; ++i) {
    const int c = t * 4 + i;
    orow[c] = (_Float16)((vals[i] - mu) * rstd * gamma[c] + beta[c]);
  }
}

// ---------------------------------------------------------------------------
// 2. fp32 -> f16 cast (weights). n is a multiple of 1024.
// ---------------------------------------------------------------------------
__global__ __launch_bounds__(256) void cast_f16_kernel(
    const float* __restrict__ in, _Float16* __restrict__ out, int n) {
  const int i = blockIdx.x * 1024 + threadIdx.x * 4;
  if (i + 3 < n) {
    const float4 f = *(const float4*)(in + i);
    out[i + 0] = (_Float16)f.x;
    out[i + 1] = (_Float16)f.y;
    out[i + 2] = (_Float16)f.z;
    out[i + 3] = (_Float16)f.w;
  }
}

// ---------------------------------------------------------------------------
// 3/5. Tiled WMMA GEMM: C[MxN] = A[MxK] * W[KxN]; 64x64 tile / block,
// 8 waves, each wave computes a 16x64 strip (4 C tiles).
// Double-buffered async-to-LDS staging overlaps HBM with WMMA.
// ---------------------------------------------------------------------------
template <bool F32OUT>
__global__ __launch_bounds__(256) void gemm_wmma_kernel(
    const _Float16* __restrict__ A, const _Float16* __restrict__ W,
    _Float16* __restrict__ Ch, float* __restrict__ Cf,
    const float* __restrict__ bias, int M, int N, int K) {
  __shared__ _Float16 As[2][64][LDSP];
  __shared__ _Float16 Ws[2][64][LDSP];

  const int tile_m = blockIdx.y * 64;
  const int tile_n = blockIdx.x * 64;
  const int t    = threadIdx.x;
  const int wave = t >> 5;
  const int lane = t & 31;

  auto stage = [&](int buf, int k0) {
#pragma unroll
    for (int it = 0; it < 2; ++it) {         // 512 16B chunks, 256 threads
      const int idx = t + it * 256;
      const int r   = idx >> 3;
      const int c   = (idx & 7) * 8;
      async_copy16(&As[buf][r][c], &A[(size_t)(tile_m + r) * K + k0 + c]);
      async_copy16(&Ws[buf][r][c], &W[(size_t)(k0 + r) * N + tile_n + c]);
    }
  };

  v8f acc[4] = {};

  stage(0, 0);
  async_wait0();
  __syncthreads();

  int cur = 0;
  for (int k0 = 0; k0 < K; k0 += 64) {
    if (k0 + 64 < K) stage(cur ^ 1, k0 + 64);   // prefetch next tile async

#pragma unroll
    for (int kk = 0; kk < 2; ++kk) {
      const v16h a = load_a_frag(&As[cur][wave * 16][kk * 32], LDSP, lane);
#pragma unroll
      for (int nt = 0; nt < 4; ++nt) {
        const v16h b = load_b_frag(&Ws[cur][kk * 32][nt * 16], LDSP, lane);
        acc[nt] = wmma_f16(a, b, acc[nt]);
      }
    }

    async_wait0();       // next buffer landed
    __syncthreads();     // everyone done reading cur + everyone's copies done
    cur ^= 1;
  }

  // C/D layout: vgpr r holds row (r + 8*(lane>=16)), col = lane&15 per tile
  const int hi8 = (lane >> 4) << 3;
  const int n0  = tile_n + (lane & 15);
#pragma unroll
  for (int nt = 0; nt < 4; ++nt) {
#pragma unroll
    for (int r = 0; r < 8; ++r) {
      const int m   = tile_m + wave * 16 + hi8 + r;
      const int col = n0 + nt * 16;
      if constexpr (F32OUT)
        Cf[(size_t)m * N + col] = acc[nt][r] + bias[col];
      else
        Ch[(size_t)m * N + col] = (_Float16)acc[nt][r];
    }
  }
}

// ---------------------------------------------------------------------------
// 4. Flash attention: block = (b, h, 64-query tile), 8 waves.
//    Q frags in registers; K/V double-buffered via async-to-LDS; P through
//    LDS for the P*V A-fragments; online softmax in f32.
//    qkv layout: [B*N, 3072] f16, cols: [0,1024)=Q, [1024,2048)=K, [2048,3072)=V.
// ---------------------------------------------------------------------------
__global__ __launch_bounds__(256) void attn_wmma_kernel(
    const _Float16* __restrict__ qkv_q,   // this stream's Q
    const _Float16* __restrict__ qkv_kv,  // other stream's K,V
    _Float16* __restrict__ out) {         // [B*N, 1024] f16
  __shared__ _Float16 Qs[64][LDSP];
  __shared__ _Float16 Ks[2][64][LDSP];
  __shared__ _Float16 Vs[2][64][LDSP];
  __shared__ _Float16 Ps[64][LDSP];

  const int qt = blockIdx.x, h = blockIdx.y, b = blockIdx.z;
  const int t = threadIdx.x, wave = t >> 5, lane = t & 31;
  const size_t rowbase = (size_t)b * NSEQ;
  const int hcol = h * DIM_HEAD;
  const int hi8  = (lane >> 4) << 3;

  auto stage_kv = [&](int buf, int kt) {
#pragma unroll
    for (int it = 0; it < 2; ++it) {
      const int idx = t + it * 256;
      const int r = idx >> 3, c = (idx & 7) * 8;
      const size_t krow = (rowbase + kt * 64 + r) * NQKV + hcol;
      async_copy16(&Ks[buf][r][c], &qkv_kv[krow + DIM + c]);
      async_copy16(&Vs[buf][r][c], &qkv_kv[krow + 2 * DIM + c]);
    }
  };

  // stage Q tile + first K/V tile asynchronously
#pragma unroll
  for (int it = 0; it < 2; ++it) {
    const int idx = t + it * 256;
    const int r = idx >> 3, c = (idx & 7) * 8;
    async_copy16(&Qs[r][c],
                 &qkv_q[(rowbase + qt * 64 + r) * NQKV + hcol + c]);
  }
  stage_kv(0, 0);
  async_wait0();
  __syncthreads();

  v16h qf[2];
  qf[0] = load_a_frag(&Qs[wave * 16][0],  LDSP, lane);
  qf[1] = load_a_frag(&Qs[wave * 16][32], LDSP, lane);

  v8f o[4] = {};
  float row_max[8], row_sum[8];
#pragma unroll
  for (int r = 0; r < 8; ++r) { row_max[r] = -3.0e38f; row_sum[r] = 0.f; }

  int cur = 0;
  for (int kt = 0; kt < NSEQ / 64; ++kt) {
    if (kt + 1 < NSEQ / 64) stage_kv(cur ^ 1, kt + 1);  // async prefetch

    // S = Q * K^T  (wave: 16 q-rows x 64 keys = 4 tiles)
    v8f s[4] = {};
#pragma unroll
    for (int kk = 0; kk < 2; ++kk) {
#pragma unroll
      for (int nt = 0; nt < 4; ++nt) {
        const v16h bk = load_bT_frag(&Ks[cur][nt * 16][kk * 32], LDSP, lane);
        s[nt] = wmma_f16(qf[kk], bk, s[nt]);
      }
    }

    // online softmax: rows live across 16-lane halves (ISA C/D layout)
#pragma unroll
    for (int r = 0; r < 8; ++r) {
      float mx = fmaxf(fmaxf(s[0][r], s[1][r]), fmaxf(s[2][r], s[3][r])) * ATT_SCALE;
#pragma unroll
      for (int m = 8; m >= 1; m >>= 1) mx = fmaxf(mx, __shfl_xor(mx, m, 32));
      const float nmax  = fmaxf(row_max[r], mx);
      const float alpha = __expf(row_max[r] - nmax);
      float psum = 0.f;
#pragma unroll
      for (int nt = 0; nt < 4; ++nt) {
        const float p = __expf(s[nt][r] * ATT_SCALE - nmax);
        s[nt][r] = p;
        psum += p;
      }
#pragma unroll
      for (int m = 8; m >= 1; m >>= 1) psum += __shfl_xor(psum, m, 32);
      row_sum[r] = row_sum[r] * alpha + psum;
      row_max[r] = nmax;
#pragma unroll
      for (int nt = 0; nt < 4; ++nt) o[nt][r] *= alpha;   // rescale O
    }

    // P -> LDS (f16), then O += P * V via WMMA
#pragma unroll
    for (int nt = 0; nt < 4; ++nt)
#pragma unroll
      for (int r = 0; r < 8; ++r)
        Ps[wave * 16 + hi8 + r][nt * 16 + (lane & 15)] = (_Float16)s[nt][r];
    __syncthreads();

#pragma unroll
    for (int kk = 0; kk < 2; ++kk) {
      const v16h pa = load_a_frag(&Ps[wave * 16][kk * 32], LDSP, lane);
#pragma unroll
      for (int nt = 0; nt < 4; ++nt) {
        const v16h vb = load_b_frag(&Vs[cur][kk * 32][nt * 16], LDSP, lane);
        o[nt] = wmma_f16(pa, vb, o[nt]);
      }
    }

    async_wait0();       // next K/V buffer landed
    __syncthreads();     // all waves done with cur K/V and P
    cur ^= 1;
  }

  // normalize and store f16 (consumed by out-proj GEMM)
#pragma unroll
  for (int r = 0; r < 8; ++r) {
    const float inv = 1.0f / row_sum[r];
    const size_t m = rowbase + qt * 64 + wave * 16 + hi8 + r;
#pragma unroll
    for (int nt = 0; nt < 4; ++nt)
      out[m * DIM + hcol + nt * 16 + (lane & 15)] = (_Float16)(o[nt][r] * inv);
  }
}

// ---------------------------------------------------------------------------
// launch
// ---------------------------------------------------------------------------
extern "C" void kernel_launch(void* const* d_in, const int* in_sizes, int n_in,
                              void* d_out, int out_size, void* d_ws, size_t ws_size,
                              hipStream_t stream) {
  (void)in_sizes; (void)n_in; (void)out_size; (void)ws_size;
  const float* x1     = (const float*)d_in[0];
  const float* x2     = (const float*)d_in[1];
  const float* ln1_g  = (const float*)d_in[2];
  const float* ln1_b  = (const float*)d_in[3];
  const float* ln2_g  = (const float*)d_in[4];
  const float* ln2_b  = (const float*)d_in[5];
  const float* w_qkv1 = (const float*)d_in[6];
  const float* w_qkv2 = (const float*)d_in[7];
  const float* w_out1 = (const float*)d_in[8];
  const float* b_out1 = (const float*)d_in[9];
  const float* w_out2 = (const float*)d_in[10];
  const float* b_out2 = (const float*)d_in[11];
  float* out = (float*)d_out;

  // workspace carve-up (f16 elements)
  _Float16* ws  = (_Float16*)d_ws;
  size_t off = 0;
  _Float16* x1n  = ws + off; off += (size_t)MROWS * DIM;        // 8192x1024
  _Float16* x2n  = ws + off; off += (size_t)MROWS * DIM;
  _Float16* wq1  = ws + off; off += (size_t)DIM * NQKV;         // 1024x3072
  _Float16* wq2  = ws + off; off += (size_t)DIM * NQKV;
  _Float16* wo1  = ws + off; off += (size_t)DIM * DIM;          // 1024x1024
  _Float16* wo2  = ws + off; off += (size_t)DIM * DIM;
  _Float16* qkv1 = ws + off; off += (size_t)MROWS * NQKV;       // 8192x3072
  _Float16* qkv2 = ws + off; off += (size_t)MROWS * NQKV;
  _Float16* ao1  = ws + off; off += (size_t)MROWS * DIM;        // 8192x1024
  _Float16* ao2  = ws + off; off += (size_t)MROWS * DIM;

  // 1. layernorm + cast
  ln_cast_kernel<<<MROWS, 256, 0, stream>>>(x1, ln1_g, ln1_b, x1n);
  ln_cast_kernel<<<MROWS, 256, 0, stream>>>(x2, ln2_g, ln2_b, x2n);

  // 2. weight casts
  cast_f16_kernel<<<(DIM * NQKV) / 1024, 256, 0, stream>>>(w_qkv1, wq1, DIM * NQKV);
  cast_f16_kernel<<<(DIM * NQKV) / 1024, 256, 0, stream>>>(w_qkv2, wq2, DIM * NQKV);
  cast_f16_kernel<<<(DIM * DIM)  / 1024, 256, 0, stream>>>(w_out1, wo1, DIM * DIM);
  cast_f16_kernel<<<(DIM * DIM)  / 1024, 256, 0, stream>>>(w_out2, wo2, DIM * DIM);

  // 3. QKV projections (f16 out)
  {
    dim3 g(NQKV / 64, MROWS / 64);
    gemm_wmma_kernel<false><<<g, 256, 0, stream>>>(x1n, wq1, qkv1, nullptr,
                                                   nullptr, MROWS, NQKV, DIM);
    gemm_wmma_kernel<false><<<g, 256, 0, stream>>>(x2n, wq2, qkv2, nullptr,
                                                   nullptr, MROWS, NQKV, DIM);
  }

  // 4. cross attention: out1 = SA(q1, k2, v2), out2 = SA(q2, k1, v1)
  {
    dim3 g(NSEQ / 64, HEADS, BATCH);
    attn_wmma_kernel<<<g, 256, 0, stream>>>(qkv1, qkv2, ao1);
    attn_wmma_kernel<<<g, 256, 0, stream>>>(qkv2, qkv1, ao2);
  }

  // 5. output projections (+bias, f32 into d_out)
  {
    dim3 g(DIM / 64, MROWS / 64);
    gemm_wmma_kernel<true><<<g, 256, 0, stream>>>(ao1, wo1, nullptr, out,
                                                  b_out1, MROWS, DIM, DIM);
    gemm_wmma_kernel<true><<<g, 256, 0, stream>>>(ao2, wo2, nullptr,
                                                  out + (size_t)MROWS * DIM,
                                                  b_out2, MROWS, DIM, DIM);
  }
}
